// UnifiedPhasorBlock_37082747633920
// MI455X (gfx1250) — compile-verified, matrix operand order checked by reading, not run on hardware
//
#include <hip/hip_runtime.h>

// ---------------------------------------------------------------------------
// UnifiedPhasorBlock for MI455X (gfx1250, wave32, WMMA).
// Rewrites phasor-binding cumsum attention as causal linear attention:
//   retrieved = tril(Qe Ke^T) V,  Qe/Ke = [amp*cos(ph) | amp*sin(ph)] (L x 128)
// All GEMMs run on v_wmma_f32_16x16x32_f16 (f16 in, f32 accumulate).
// Workspace need: ~5.9 MB.
// ---------------------------------------------------------------------------

#define LSEQ 1024
#define DDIM 256
#define KDIM 64
#define NB   2
#define KE   128   // 2*K (real|imag concatenated)

typedef __attribute__((ext_vector_type(16))) _Float16 v16h;
typedef __attribute__((ext_vector_type(8)))  _Float16 v8h;
typedef __attribute__((ext_vector_type(8)))  float    v8f;

// ---- WMMA fragment helpers -------------------------------------------------
// A-frag (16x32 f16): lane&15 = M row; element e <-> K = (e/8)*16+(lane>>4)*8+e%8
// B-frag (32x16 f16): same pattern over B^T stored row-major (lane&15 = N row)
__device__ __forceinline__ v16h load_frag(const _Float16* base, int stride, int lane) {
  const _Float16* p = base + (size_t)(lane & 15) * stride + ((lane >> 4) * 8);
  v8h lo = *(const v8h*)(p);        // K = (lane>>4)*8 + 0..7
  v8h hi = *(const v8h*)(p + 16);   // K = 16 + (lane>>4)*8 + 0..7
  v16h out;
#pragma unroll
  for (int i = 0; i < 8; ++i) { out[i] = lo[i]; out[i + 8] = hi[i]; }
  return out;
}

__device__ __forceinline__ v8f wmma16(v16h a, v16h b, v8f c) {
  return __builtin_amdgcn_wmma_f32_16x16x32_f16(false, a, false, b, (short)0, c,
                                                false, false);
}

// ---- scalar math -----------------------------------------------------------
__device__ __forceinline__ float gelu_exact(float x) {
  return 0.5f * x * (1.0f + erff(x * 0.70710678118654752f));
}
__device__ __forceinline__ float softplus_f(float x) {
  return (x > 20.f) ? x : log1pf(expf(x));
}

// ---- kernel 1: weight convert + transpose to f16 ---------------------------
__global__ void convert_transpose(const float* __restrict__ in,
                                  _Float16* __restrict__ out, int Kr, int Nc) {
  int idx = blockIdx.x * blockDim.x + threadIdx.x;
  if (idx >= Kr * Nc) return;
  int n = idx / Kr, k = idx % Kr;           // out[n][k] = in[k][n]
  out[(size_t)n * Kr + k] = (_Float16)in[(size_t)k * Nc + n];
}

// ---- kernel 2: encoder (k-phasors / q-phasors / V) -------------------------
// grid.x = NB * 3 * (LSEQ/16); one wave per 16-row tile.
__global__ __launch_bounds__(32) void encode_kernel(
    const float* __restrict__ x, const float* __restrict__ pos_k,
    const float* __restrict__ pos_q,
    const float* __restrict__ b1k, const float* __restrict__ b2k,
    const float* __restrict__ bak,
    const float* __restrict__ b1q, const float* __restrict__ b2q,
    const float* __restrict__ baq, const float* __restrict__ bvv,
    const _Float16* __restrict__ w1k_t, const _Float16* __restrict__ w2k_t,
    const _Float16* __restrict__ wak_t,
    const _Float16* __restrict__ w1q_t, const _Float16* __restrict__ w2q_t,
    const _Float16* __restrict__ waq_t, const _Float16* __restrict__ wv_t,
    _Float16* __restrict__ Ek, _Float16* __restrict__ Eq,
    _Float16* __restrict__ Vt) {
  __shared__ __align__(16) _Float16 xa_h[16 * DDIM];
  __shared__ __align__(16) _Float16 h_h[16 * DDIM];
  __shared__ float amp_s[16 * KDIM];

  const int lane = threadIdx.x;
  const int tile = blockIdx.x % (LSEQ / 16);
  const int side = (blockIdx.x / (LSEQ / 16)) % 3;     // 0=k, 1=q, 2=V
  const int b    = blockIdx.x / (3 * (LSEQ / 16));
  const int l0   = tile * 16;
  const int M    = (lane >> 4) * 8;   // C-frag row base
  const int N    = lane & 15;         // C-frag column

  const float* pos = (side == 0) ? pos_k : (side == 1) ? pos_q : nullptr;

  // xa = x (+ pos) -> LDS f16, 16 x 256
  for (int i = 0; i < (16 * DDIM) / 32; ++i) {
    int idx = i * 32 + lane;
    int r = idx >> 8, c = idx & 255;
    float v = x[((size_t)(b * LSEQ + l0 + r)) * DDIM + c];
    if (pos) v += pos[(size_t)(l0 + r) * DDIM + c];
    xa_h[idx] = (_Float16)v;
  }

  if (side == 2) {  // V = x@wv + bv, stored transposed (D x L) f16
    for (int nt = 0; nt < 16; ++nt) {
      v8f acc = {};
#pragma unroll
      for (int kc = 0; kc < 8; ++kc) {
        v16h a  = load_frag(xa_h + kc * 32, DDIM, lane);
        v16h bm = load_frag(wv_t + (size_t)(nt * 16) * DDIM + kc * 32, DDIM, lane);
        acc = wmma16(a, bm, acc);
      }
      int d = nt * 16 + N;
      float bias = bvv[d];
#pragma unroll
      for (int r = 0; r < 8; ++r)
        Vt[((size_t)b * DDIM + d) * LSEQ + (l0 + r + M)] = (_Float16)(acc[r] + bias);
    }
    return;
  }

  const float* b1 = side ? b1q : b1k;
  const float* b2 = side ? b2q : b2k;
  const float* ba = side ? baq : bak;
  const _Float16* w1t = side ? w1q_t : w1k_t;
  const _Float16* w2t = side ? w2q_t : w2k_t;
  const _Float16* wat = side ? waq_t : wak_t;
  _Float16* E = side ? Eq : Ek;

  // amp = softplus(xa@wa + ba) + 0.1  -> LDS
  for (int nt = 0; nt < 4; ++nt) {
    v8f acc = {};
#pragma unroll
    for (int kc = 0; kc < 8; ++kc) {
      v16h a  = load_frag(xa_h + kc * 32, DDIM, lane);
      v16h bm = load_frag(wat + (size_t)(nt * 16) * DDIM + kc * 32, DDIM, lane);
      acc = wmma16(a, bm, acc);
    }
    int c = nt * 16 + N;
    float bias = ba[c];
#pragma unroll
    for (int r = 0; r < 8; ++r)
      amp_s[(r + M) * KDIM + c] = softplus_f(acc[r] + bias) + 0.1f;
  }

  // h = gelu(xa@w1 + b1) -> LDS f16
  for (int nt = 0; nt < 16; ++nt) {
    v8f acc = {};
#pragma unroll
    for (int kc = 0; kc < 8; ++kc) {
      v16h a  = load_frag(xa_h + kc * 32, DDIM, lane);
      v16h bm = load_frag(w1t + (size_t)(nt * 16) * DDIM + kc * 32, DDIM, lane);
      acc = wmma16(a, bm, acc);
    }
    int c = nt * 16 + N;
    float bias = b1[c];
#pragma unroll
    for (int r = 0; r < 8; ++r)
      h_h[(r + M) * DDIM + c] = (_Float16)gelu_exact(acc[r] + bias);
  }

  // phase = tanh(h@w2 + b2)*pi ; E = [amp*cos | amp*sin] (f16)
  for (int nt = 0; nt < 4; ++nt) {
    v8f acc = {};
#pragma unroll
    for (int kc = 0; kc < 8; ++kc) {
      v16h a  = load_frag(h_h + kc * 32, DDIM, lane);
      v16h bm = load_frag(w2t + (size_t)(nt * 16) * DDIM + kc * 32, DDIM, lane);
      acc = wmma16(a, bm, acc);
    }
    int c = nt * 16 + N;
    float bias = b2[c];
#pragma unroll
    for (int r = 0; r < 8; ++r) {
      float ph = tanhf(acc[r] + bias) * 3.14159265358979f;
      float am = amp_s[(r + M) * KDIM + c];
      float sn, cs;
      sincosf(ph, &sn, &cs);
      size_t base = ((size_t)(b * LSEQ + l0 + r + M)) * KE + c;
      E[base]        = (_Float16)(am * cs);
      E[base + KDIM] = (_Float16)(am * sn);
    }
  }
}

// ---- kernel 3: causal linear attention ------------------------------------
// grid.x = NB * (LSEQ/16); 4 waves per block, each wave owns 64 of D.
__global__ __launch_bounds__(128) void attn_kernel(
    const _Float16* __restrict__ Ek, const _Float16* __restrict__ Eq,
    const _Float16* __restrict__ Vt, float* __restrict__ ret) {
  __shared__ __align__(16) _Float16 s_tile[4][16 * 32];
  const int lane = threadIdx.x & 31;
  const int wave = threadIdx.x >> 5;
  const int tile = blockIdx.x % (LSEQ / 16);
  const int b    = blockIdx.x / (LSEQ / 16);
  const int l0   = tile * 16;
  const int M    = (lane >> 4) * 8;
  const int N    = lane & 15;

  v16h aq[4];
#pragma unroll
  for (int c = 0; c < 4; ++c)
    aq[c] = load_frag(Eq + ((size_t)(b * LSEQ + l0)) * KE + c * 32, KE, lane);

  v8f accd[4] = {};

  const int nch = tile / 2 + 1;  // chunks of 32 keys
  for (int kt = 0; kt < nch; ++kt) {
#pragma unroll
    for (int h = 0; h < 2; ++h) {
      int t0 = kt * 32 + h * 16;
      if (t0 > l0) {  // entirely future: zero this half of the S strip
#pragma unroll
        for (int r = 0; r < 8; ++r)
          s_tile[wave][(r + M) * 32 + h * 16 + N] = (_Float16)0.f;
      } else {
        v8f s = {};
#pragma unroll
        for (int c = 0; c < 4; ++c) {
          v16h bk = load_frag(Ek + ((size_t)(b * LSEQ + t0)) * KE + c * 32, KE, lane);
          s = wmma16(aq[c], bk, s);
        }
        if (t0 == l0) {  // diagonal tile: keep t <= l  (N <= M_row)
#pragma unroll
          for (int r = 0; r < 8; ++r)
            if (N > r + M) s[r] = 0.f;
        }
#pragma unroll
        for (int r = 0; r < 8; ++r)
          s_tile[wave][(r + M) * 32 + h * 16 + N] = (_Float16)s[r];
      }
    }
    // retrieved += S(16x32) @ V(32 x 64-slice), via V^T
    v16h a = load_frag(&s_tile[wave][0], 32, lane);
#pragma unroll
    for (int nt = 0; nt < 4; ++nt) {
      int d0 = wave * 64 + nt * 16;
      v16h bm = load_frag(Vt + ((size_t)b * DDIM + d0) * LSEQ + kt * 32, LSEQ, lane);
      accd[nt] = wmma16(a, bm, accd[nt]);
    }
  }

#pragma unroll
  for (int nt = 0; nt < 4; ++nt)
#pragma unroll
    for (int r = 0; r < 8; ++r) {
      int l = l0 + r + M;
      float nrm = sqrtf((float)(l + 1) * (float)KDIM);
      ret[((size_t)(b * LSEQ + l)) * DDIM + wave * 64 + nt * 16 + N] =
          accd[nt][r] / nrm;
    }
}

// ---- kernel 4: LayerNorm (wave per row) ------------------------------------
__global__ __launch_bounds__(32) void ln_kernel(const float* __restrict__ ret,
                                                const float* __restrict__ g,
                                                const float* __restrict__ bt,
                                                _Float16* __restrict__ rn) {
  const int row = blockIdx.x;
  const int lane = threadIdx.x;
  float v[8], s = 0.f, s2 = 0.f;
#pragma unroll
  for (int i = 0; i < 8; ++i) {
    v[i] = ret[(size_t)row * DDIM + lane * 8 + i];
    s += v[i];
    s2 += v[i] * v[i];
  }
#pragma unroll
  for (int o = 16; o > 0; o >>= 1) {
    s  += __shfl_xor(s, o, 32);
    s2 += __shfl_xor(s2, o, 32);
  }
  float mu   = s * (1.f / DDIM);
  float var  = s2 * (1.f / DDIM) - mu * mu;
  float rstd = rsqrtf(var + 1e-5f);
#pragma unroll
  for (int i = 0; i < 8; ++i) {
    int c = lane * 8 + i;
    rn[(size_t)row * DDIM + c] = (_Float16)((v[i] - mu) * rstd * g[c] + bt[c]);
  }
}

// ---- kernel 5: out = x + rn@wo + bo ---------------------------------------
// grid.x = NB*LSEQ/16; one wave per 16-row tile.
__global__ __launch_bounds__(32) void final_kernel(
    const _Float16* __restrict__ rn, const _Float16* __restrict__ wo_t,
    const float* __restrict__ x, const float* __restrict__ bo,
    float* __restrict__ out) {
  const int lane = threadIdx.x;
  const int row0 = blockIdx.x * 16;
  const int M = (lane >> 4) * 8;
  const int N = lane & 15;
  v16h afr[8];
#pragma unroll
  for (int kc = 0; kc < 8; ++kc)
    afr[kc] = load_frag(rn + (size_t)row0 * DDIM + kc * 32, DDIM, lane);
  for (int nt = 0; nt < 16; ++nt) {
    v8f acc = {};
#pragma unroll
    for (int kc = 0; kc < 8; ++kc) {
      v16h bm = load_frag(wo_t + (size_t)(nt * 16) * DDIM + kc * 32, DDIM, lane);
      acc = wmma16(afr[kc], bm, acc);
    }
    int c = nt * 16 + N;
    float bias = bo[c];
#pragma unroll
    for (int r = 0; r < 8; ++r) {
      size_t idx = (size_t)(row0 + r + M) * DDIM + c;
      out[idx] = x[idx] + acc[r] + bias;
    }
  }
}

// ---- host glue -------------------------------------------------------------
extern "C" void kernel_launch(void* const* d_in, const int* in_sizes, int n_in,
                              void* d_out, int out_size, void* d_ws,
                              size_t ws_size, hipStream_t stream) {
  const float* x    = (const float*)d_in[0];
  const float* posk = (const float*)d_in[1];
  const float* w1k  = (const float*)d_in[2];
  const float* b1k  = (const float*)d_in[3];
  const float* w2k  = (const float*)d_in[4];
  const float* b2k  = (const float*)d_in[5];
  const float* wak  = (const float*)d_in[6];
  const float* bak  = (const float*)d_in[7];
  const float* posq = (const float*)d_in[8];
  const float* w1q  = (const float*)d_in[9];
  const float* b1q  = (const float*)d_in[10];
  const float* w2q  = (const float*)d_in[11];
  const float* b2q  = (const float*)d_in[12];
  const float* waq  = (const float*)d_in[13];
  const float* baq  = (const float*)d_in[14];
  const float* wv   = (const float*)d_in[15];
  const float* bv   = (const float*)d_in[16];
  const float* lng  = (const float*)d_in[17];
  const float* lnb  = (const float*)d_in[18];
  const float* wo   = (const float*)d_in[19];
  const float* bo   = (const float*)d_in[20];

  char* ws = (char*)d_ws;
  // byte offsets (all 256-B aligned); total ~5.9 MB
  _Float16* w1k_t = (_Float16*)(ws + 0);
  _Float16* w2k_t = (_Float16*)(ws + 131072);
  _Float16* wak_t = (_Float16*)(ws + 163840);
  _Float16* w1q_t = (_Float16*)(ws + 196608);
  _Float16* w2q_t = (_Float16*)(ws + 327680);
  _Float16* waq_t = (_Float16*)(ws + 360448);
  _Float16* wv_t  = (_Float16*)(ws + 393216);
  _Float16* wo_t  = (_Float16*)(ws + 524288);
  _Float16* Ek    = (_Float16*)(ws + 655360);   // NB*L*128 f16
  _Float16* Eq    = (_Float16*)(ws + 1179648);
  _Float16* Vt    = (_Float16*)(ws + 1703936);  // NB*D*L f16
  _Float16* rn    = (_Float16*)(ws + 2752512);  // NB*L*D f16
  float*    ret   = (float*)   (ws + 3801088);  // NB*L*D f32

  // 1) weights -> f16 transposed
  convert_transpose<<<(DDIM * DDIM + 255) / 256, 256, 0, stream>>>(w1k, w1k_t, DDIM, DDIM);
  convert_transpose<<<(DDIM * KDIM + 255) / 256, 256, 0, stream>>>(w2k, w2k_t, DDIM, KDIM);
  convert_transpose<<<(DDIM * KDIM + 255) / 256, 256, 0, stream>>>(wak, wak_t, DDIM, KDIM);
  convert_transpose<<<(DDIM * DDIM + 255) / 256, 256, 0, stream>>>(w1q, w1q_t, DDIM, DDIM);
  convert_transpose<<<(DDIM * KDIM + 255) / 256, 256, 0, stream>>>(w2q, w2q_t, DDIM, KDIM);
  convert_transpose<<<(DDIM * KDIM + 255) / 256, 256, 0, stream>>>(waq, waq_t, DDIM, KDIM);
  convert_transpose<<<(DDIM * DDIM + 255) / 256, 256, 0, stream>>>(wv, wv_t, DDIM, DDIM);
  convert_transpose<<<(DDIM * DDIM + 255) / 256, 256, 0, stream>>>(wo, wo_t, DDIM, DDIM);

  // 2) encode phasors + V
  encode_kernel<<<NB * 3 * (LSEQ / 16), 32, 0, stream>>>(
      x, posk, posq, b1k, b2k, bak, b1q, b2q, baq, bv,
      w1k_t, w2k_t, wak_t, w1q_t, w2q_t, waq_t, wv_t, Ek, Eq, Vt);

  // 3) causal linear attention
  attn_kernel<<<NB * (LSEQ / 16), 128, 0, stream>>>(Ek, Eq, Vt, ret);

  // 4) layernorm
  ln_kernel<<<NB * LSEQ, 32, 0, stream>>>(ret, lng, lnb, rn);

  // 5) residual + output projection
  final_kernel<<<NB * (LSEQ / 16), 32, 0, stream>>>(rn, wo_t, x, bo, (float*)d_out);
}
